// HeteroGAT_47828755808348
// MI455X (gfx1250) — matrix operand order repaired and checked by reading, not compiled, not run on hardware
//
#include <hip/hip_runtime.h>

typedef __attribute__((ext_vector_type(16))) _Float16 v16h;
typedef __attribute__((ext_vector_type(8)))  _Float16 v8h;
typedef __attribute__((ext_vector_type(8)))  float    v8f;
typedef __attribute__((ext_vector_type(4)))  unsigned u32x4;
typedef __attribute__((ext_vector_type(8)))  int      i32x8;
typedef __attribute__((ext_vector_type(4)))  int      i32x4;

#if defined(__AMDGCN__) && __has_builtin(__builtin_amdgcn_tensor_load_to_lds) && \
    __has_builtin(__builtin_amdgcn_s_wait_tensorcnt)
#define USE_TDM 1
#else
#define USE_TDM 0
#endif

#if USE_TDM
// 2D tile DMA: global (f16, row stride stride_elems) -> LDS at lds_byte.
// tensor_dim set to remaining extent => TDM zero-fills K-tail / M-tail.
// pad: after every 16 DWORDs (one 32-half row) insert 4 DWORDs -> LDS row
// stride = 40 halves, matching As[.][40]/Bs[.][40].
__device__ __forceinline__ void tdm2d(unsigned lds_byte, const _Float16* g,
                                      int rows_rem, int cols_rem,
                                      int stride_elems, int tile_rows) {
  unsigned long long ga = (unsigned long long)g;
  u32x4 g0;
  g0[0] = 1u;                                   // count=1 (valid descriptor)
  g0[1] = lds_byte;                             // lds_addr
  g0[2] = (unsigned)ga;                         // global_addr lo
  g0[3] = (unsigned)((ga >> 32) & 0x01FFFFFFull) | (2u << 30);  // type=2
  unsigned d0 = (unsigned)cols_rem, d1 = (unsigned)rows_rem;
  unsigned long long st = (unsigned long long)(unsigned)stride_elems;
  i32x8 g1;
  g1[0] = (int)((1u << 16)      // data_size = 2 bytes
              | (1u << 20)      // pad_enable
              | (3u << 22)      // pad_interval: 16 DWORDs
              | (3u << 25));    // pad_amount:   4 DWORDs
  g1[1] = (int)((d0 & 0xFFFFu) << 16);                       // tensor_dim0 lo
  g1[2] = (int)((d0 >> 16) | ((d1 & 0xFFFFu) << 16));        // dim0 hi | dim1 lo
  g1[3] = (int)((d1 >> 16) | (32u << 16));                   // dim1 hi | tile_dim0=32
  g1[4] = (int)(tile_rows & 0xFFFF);                         // tile_dim1 (tile_dim2=0)
  g1[5] = (int)(st & 0xFFFFFFFFull);                         // tensor_dim0_stride lo
  g1[6] = (int)((st >> 32) & 0xFFFFull);                     // stride hi (dim1_stride=0)
  g1[7] = 0;
  i32x4 z4 = {0, 0, 0, 0};
  i32x8 z8 = {0, 0, 0, 0, 0, 0, 0, 0};
  // 6-arg form (amdgpu-toolchain / clang-23): groups 2..4 unused for 2D tiles
  __builtin_amdgcn_tensor_load_to_lds(g0, g1, z4, z4, z8, 0);
}
#endif

// ---------------------------------------------------------------------------
// WMMA GEMM: C[M,N] (+)= A[M,K] * Bt[N,K]^T (+ bias[N])
// A: f16 row-major [M][K]; Bt: f16 [N][K] (pre-transposed weights).
// Block tile 32x64, 8 waves, one 16x16 WMMA tile per wave, f32 accumulate.
// Tiles staged into LDS by the Tensor Data Mover (wave 0), stride-40 rows.
// ---------------------------------------------------------------------------
template<bool ACCUM, bool BIAS>
__global__ __launch_bounds__(256)
void k_wmma_gemm(const _Float16* __restrict__ A, const _Float16* __restrict__ Bt,
                 const float* __restrict__ bias, float* __restrict__ C,
                 int M, int N, int K) {
  __shared__ __align__(16) _Float16 As[32][40];   // [m][k]
  __shared__ __align__(16) _Float16 Bs[64][40];   // [n][k]
  const int tid  = threadIdx.x;
  const int lane = tid & 31;
  const int wid  = tid >> 5;
  const int wrow = (wid >> 2) << 4;
  const int wcol = (wid & 3) << 4;
  const int bm = blockIdx.x << 5;
  const int bn = blockIdx.y << 6;
  v8f acc = {0.f,0.f,0.f,0.f,0.f,0.f,0.f,0.f};
#if USE_TDM
  const unsigned ldsA = (unsigned)(unsigned long long)&As[0][0];
  const unsigned ldsB = (unsigned)(unsigned long long)&Bs[0][0];
#endif

  for (int k0 = 0; k0 < K; k0 += 32) {
    __syncthreads();                        // previous tiles consumed
#if USE_TDM
    if (wid == 0) {
      tdm2d(ldsA, A + (long long)bm * K + k0, M - bm, K - k0, K, 32);
      tdm2d(ldsB, Bt + (long long)bn * K + k0, N - bn, K - k0, K, 64);
      __builtin_amdgcn_s_wait_tensorcnt(0);
    }
#else
    for (int i = tid; i < 32*32; i += 256) {
      int r = i >> 5, c = i & 31;
      int gr = bm + r, gk = k0 + c;
      As[r][c] = (gr < M && gk < K) ? A[(long long)gr * K + gk] : (_Float16)0.f;
    }
    for (int i = tid; i < 32*64; i += 256) {
      int nn = i >> 5, kk = i & 31;
      int gk = k0 + kk;
      Bs[nn][kk] = (gk < K) ? Bt[(long long)(bn + nn) * K + gk] : (_Float16)0.f;
    }
#endif
    __syncthreads();

    // A fragment: lanes 0-15 rows m, K = {kb..kb+7, kb+16..kb+23}, kb=(lane>>4)*8
    const int am  = wrow + (lane & 15);
    const int akb = (lane >> 4) << 3;
    v8h a0 = *(const v8h*)&As[am][akb];
    v8h a1 = *(const v8h*)&As[am][akb + 16];
    // B fragment: col n = lane&15, K = {kb..kb+7, kb+16..kb+23}, kb=(lane>>4)*8
    const int bnn = wcol + (lane & 15);
    const int bkb = (lane >> 4) << 3;
    v8h b0 = *(const v8h*)&Bs[bnn][bkb];
    v8h b1 = *(const v8h*)&Bs[bnn][bkb + 16];
    v16h av, bv;
    #pragma unroll
    for (int i = 0; i < 8; ++i) {
      av[i] = a0[i]; av[i+8] = a1[i];
      bv[i] = b0[i]; bv[i+8] = b1[i];
    }
    acc = __builtin_amdgcn_wmma_f32_16x16x32_f16(
        false, av, false, bv, (short)0, acc, false, false);
  }

  const int cn = bn + wcol + (lane & 15);
  #pragma unroll
  for (int j = 0; j < 8; ++j) {
    int cm = bm + wrow + ((lane >> 4) << 3) + j;
    if (cm < M) {
      float v = acc[j];
      if (BIAS) v += bias[cn];
      long long idx = (long long)cm * N + cn;
      if (ACCUM) C[idx] += v; else C[idx] = v;
    }
  }
}

// ---------------------------------------------------------------------------
// LayerNorm, one wave per row; fp32 in, f16 out (feeds WMMA A / attn gemv)
// ---------------------------------------------------------------------------
template<int D>
__global__ __launch_bounds__(256)
void k_layernorm(const float* __restrict__ x, const float* __restrict__ g,
                 const float* __restrict__ b, _Float16* __restrict__ y, int M) {
  int row  = blockIdx.x * 8 + (threadIdx.x >> 5);
  int lane = threadIdx.x & 31;
  if (row >= M) return;
  constexpr int CNT = D / 32;
  float v[CNT];
  const float* xr = x + (long long)row * D;
  float s = 0.f;
  #pragma unroll
  for (int i = 0; i < CNT; ++i) { v[i] = xr[lane + i*32]; s += v[i]; }
  #pragma unroll
  for (int o = 16; o > 0; o >>= 1) s += __shfl_xor(s, o, 32);
  float mu = s * (1.f / D);
  float var = 0.f;
  #pragma unroll
  for (int i = 0; i < CNT; ++i) { float d = v[i] - mu; var += d * d; }
  #pragma unroll
  for (int o = 16; o > 0; o >>= 1) var += __shfl_xor(var, o, 32);
  float rs = rsqrtf(var * (1.f / D) + 1e-5f);
  _Float16* yr = y + (long long)row * D;
  #pragma unroll
  for (int i = 0; i < CNT; ++i) {
    int c = lane + i*32;
    yr[c] = (_Float16)((v[i] - mu) * rs * g[c] + b[c]);
  }
}

// elementwise f32 -> f16
__global__ void k_cvt_f16(const float* __restrict__ x, _Float16* __restrict__ y,
                          long long n) {
  long long i = (long long)blockIdx.x * 256 + threadIdx.x;
  if (i < n) y[i] = (_Float16)x[i];
}

// Wt[n][k] = (f16) W[k][n]
__global__ void k_transpose_cvt(const float* __restrict__ W, _Float16* __restrict__ Wt,
                                int K, int N) {
  long long i = (long long)blockIdx.x * 256 + threadIdx.x;
  if (i >= (long long)K * N) return;
  int n = (int)(i / K), k = (int)(i % K);
  Wt[i] = (_Float16)W[(long long)k * N + n];
}

// wf[k,h] = sum_c W[k, h*64+c] * att[h*64+c]
__global__ void k_fold_att(const float* __restrict__ W, const float* __restrict__ att,
                           float* __restrict__ wf, int K) {
  int i = blockIdx.x * 256 + threadIdx.x;
  if (i >= K * 4) return;
  int k = i >> 2, h = i & 3;
  const float* wr = W + (long long)k * 256 + h * 64;
  const float* ar = att + h * 64;
  float s = 0.f;
  for (int c = 0; c < 64; ++c) s += wr[c] * ar[c];
  wf[k*4 + h] = s;
}

// out[M,4] = x16[M,K] @ wf[K,4], one wave per row
__global__ __launch_bounds__(256)
void k_attn_gemv(const _Float16* __restrict__ x, const float* __restrict__ wf,
                 float* __restrict__ out, int M, int K) {
  int row  = blockIdx.x * 8 + (threadIdx.x >> 5);
  int lane = threadIdx.x & 31;
  if (row >= M) return;
  const _Float16* xr = x + (long long)row * K;
  float a0=0.f, a1=0.f, a2=0.f, a3=0.f;
  for (int k = lane; k < K; k += 32) {
    float xv = (float)xr[k];
    a0 += xv * wf[k*4+0]; a1 += xv * wf[k*4+1];
    a2 += xv * wf[k*4+2]; a3 += xv * wf[k*4+3];
  }
  #pragma unroll
  for (int o = 16; o > 0; o >>= 1) {
    a0 += __shfl_xor(a0, o, 32); a1 += __shfl_xor(a1, o, 32);
    a2 += __shfl_xor(a2, o, 32); a3 += __shfl_xor(a3, o, 32);
  }
  if (lane == 0) {
    float* p = out + (long long)row * 4;
    p[0] = a0; p[1] = a1; p[2] = a2; p[3] = a3;
  }
}

// Edge pass 1: logit = leaky_relu(a_s[src]+a_d[dst]); segment max via
// order-preserving uint encoding + atomicMax
__global__ void k_edge_logit_max(const int* __restrict__ src, const int* __restrict__ dst,
                                 int E, int nloop, const float* __restrict__ a_s,
                                 const float* __restrict__ a_d, float* __restrict__ elog,
                                 unsigned* __restrict__ mEnc) {
  int e = blockIdx.x * 256 + threadIdx.x;
  int Et = E + nloop;
  if (e >= Et) return;
  int s = (e < E) ? src[e] : (e - E);
  int d = (e < E) ? dst[e] : (e - E);
  #pragma unroll
  for (int h = 0; h < 4; ++h) {
    float l = a_s[(long long)s*4+h] + a_d[(long long)d*4+h];
    l = (l > 0.f) ? l : 0.2f * l;
    elog[(long long)e*4+h] = l;
    unsigned bits = __float_as_uint(l);
    unsigned enc  = (bits & 0x80000000u) ? ~bits : (bits | 0x80000000u);
    atomicMax(&mEnc[(long long)d*4+h], enc);
  }
}

// Edge pass 2: e = exp(logit - max[dst]); denom[dst] += e
__global__ void k_edge_exp_sum(const int* __restrict__ dst, int E, int nloop,
                               float* __restrict__ elog, const unsigned* __restrict__ mEnc,
                               float* __restrict__ denom) {
  int e = blockIdx.x * 256 + threadIdx.x;
  int Et = E + nloop;
  if (e >= Et) return;
  int d = (e < E) ? dst[e] : (e - E);
  #pragma unroll
  for (int h = 0; h < 4; ++h) {
    unsigned enc  = mEnc[(long long)d*4+h];
    unsigned bits = (enc & 0x80000000u) ? (enc & 0x7FFFFFFFu) : ~enc;
    float m  = __uint_as_float(bits);
    float ex = __expf(elog[(long long)e*4+h] - m);
    elog[(long long)e*4+h] = ex;
    atomicAdd(&denom[(long long)d*4+h], ex);
  }
}

// Edge pass 3: out[dst,:] += hs[src,:] * alpha ; one wave per edge, 8 f32/lane
__global__ __launch_bounds__(256)
void k_edge_scatter(const int* __restrict__ src, const int* __restrict__ dst,
                    int E, int nloop, const float* __restrict__ hs,
                    const float* __restrict__ elog, const float* __restrict__ denom,
                    float* __restrict__ out) {
  int e    = blockIdx.x * 8 + (threadIdx.x >> 5);
  int lane = threadIdx.x & 31;
  int Et = E + nloop;
  if (e >= Et) return;
  int s = (e < E) ? src[e] : (e - E);
  int d = (e < E) ? dst[e] : (e - E);
  float alpha[4];
  #pragma unroll
  for (int h = 0; h < 4; ++h)
    alpha[h] = elog[(long long)e*4+h] / denom[(long long)d*4+h];
  const float* hsr = hs + (long long)s * 256;
  float* outr = out + (long long)d * 256;
  #pragma unroll
  for (int t = 0; t < 8; ++t) {
    int idx = lane + t*32;              // idx>>6 == t>>1 (head index)
    atomicAdd(&outr[idx], hsr[idx] * alpha[t >> 1]);
  }
}

// -------- small utility kernels --------
__global__ void k_fill_u32(unsigned* p, long long n, unsigned v) {
  long long i = (long long)blockIdx.x * 256 + threadIdx.x;
  if (i < n) p[i] = v;
}
__global__ void k_fill_f32(float* p, long long n, float v) {
  long long i = (long long)blockIdx.x * 256 + threadIdx.x;
  if (i < n) p[i] = v;
}
__global__ void k_rows_bias2(float* __restrict__ out, const float* __restrict__ b1,
                             const float* __restrict__ b2, long long n) {
  long long i = (long long)blockIdx.x * 256 + threadIdx.x;
  if (i < n) { int c = (int)(i & 255); out[i] = b1[c] + b2[c]; }
}
// in-place: out[i] += b1[c] + b2[c]  (layer-1 identity residual init)
__global__ void k_add_bias2(float* out, const float* __restrict__ b1,
                            const float* __restrict__ b2, long long n) {
  long long i = (long long)blockIdx.x * 256 + threadIdx.x;
  if (i < n) { int c = (int)(i & 255); out[i] += b1[c] + b2[c]; }
}
__global__ void k_elu(const float* __restrict__ x, float* __restrict__ y, long long n) {
  long long i = (long long)blockIdx.x * 256 + threadIdx.x;
  if (i < n) { float v = x[i]; y[i] = (v > 0.f) ? v : (__expf(v) - 1.f); }
}

// fused ELU + [256 x 2] classifier, one wave per row
__global__ __launch_bounds__(256)
void k_cls(const float* __restrict__ agg, const float* __restrict__ Wc,
           const float* __restrict__ bc, float* __restrict__ out, int M) {
  int row  = blockIdx.x * 8 + (threadIdx.x >> 5);
  int lane = threadIdx.x & 31;
  if (row >= M) return;
  const float* xr = agg + (long long)row * 256;
  float s0 = 0.f, s1 = 0.f;
  #pragma unroll
  for (int t = 0; t < 8; ++t) {
    int k = lane + t*32;
    float v = xr[k];
    v = (v > 0.f) ? v : (__expf(v) - 1.f);
    s0 += v * Wc[k*2+0];
    s1 += v * Wc[k*2+1];
  }
  #pragma unroll
  for (int o = 16; o > 0; o >>= 1) { s0 += __shfl_xor(s0, o, 32); s1 += __shfl_xor(s1, o, 32); }
  if (lane == 0) {
    out[(long long)row*2+0] = s0 + bc[0];
    out[(long long)row*2+1] = s1 + bc[1];
  }
}

// ---------------------------------------------------------------------------
extern "C" void kernel_launch(void* const* d_in, const int* in_sizes, int n_in,
                              void* d_out, int out_size, void* d_ws, size_t ws_size,
                              hipStream_t stream) {
  (void)in_sizes; (void)n_in; (void)out_size; (void)ws_size;
  const long long NT = 100000, NA = 150000;
  const int E_TT = 300000, E_AA = 400000, E_X = 300000;

  const float* x_tx     = (const float*)d_in[0];
  const float* x_addr   = (const float*)d_in[1];
  const int*   edge_tt  = (const int*)d_in[2];
  const int*   edge_aa  = (const int*)d_in[3];
  const int*   e_at_src = (const int*)d_in[4];
  const int*   e_at_dst = (const int*)d_in[5];
  const int*   e_ta_src = (const int*)d_in[6];
  const int*   e_ta_dst = (const int*)d_in[7];
  int p = 8;
  const float* proj_tx_W   = (const float*)d_in[p++];
  const float* proj_tx_b   = (const float*)d_in[p++];
  const float* proj_addr_W = (const float*)d_in[p++];
  const float* proj_addr_b = (const float*)d_in[p++];
  const float* res_tx_W    = (const float*)d_in[p++];
  const float* res_tx_b    = (const float*)d_in[p++];
  const float* res_addr_W  = (const float*)d_in[p++];
  const float* res_addr_b  = (const float*)d_in[p++];
  const float* ln0_tx_g    = (const float*)d_in[p++];
  const float* ln0_tx_b    = (const float*)d_in[p++];
  const float* ln0_addr_g  = (const float*)d_in[p++];
  const float* ln0_addr_b  = (const float*)d_in[p++];
  const float* ln1_tx_g    = (const float*)d_in[p++];
  const float* ln1_tx_b    = (const float*)d_in[p++];
  const float* ln1_addr_g  = (const float*)d_in[p++];
  const float* ln1_addr_b  = (const float*)d_in[p++];
  struct Gat { const float *W, *as, *ad, *bias; };
  Gat gp[6];   // tt0, aa0, at0, ta0, tt1, at1 (params dict insertion order)
  for (int i = 0; i < 6; ++i) {
    gp[i].W    = (const float*)d_in[p++];
    gp[i].as   = (const float*)d_in[p++];
    gp[i].ad   = (const float*)d_in[p++];
    gp[i].bias = (const float*)d_in[p++];
  }
  const Gat &tt0 = gp[0], &aa0 = gp[1], &at0 = gp[2], &ta0 = gp[3],
            &tt1 = gp[4], &at1 = gp[5];
  const float* cls_W = (const float*)d_in[p++];
  const float* cls_b = (const float*)d_in[p++];

  // ---- workspace: fp32 region ----
  float* ws = (float*)d_ws;
  long long o = 0;
  float* h_tx    = ws + o; o += NT * 64;
  float* h_addr  = ws + o; o += NA * 64;
  float* agg_tx  = ws + o; o += NT * 256;
  float* agg_ad  = ws + o; o += NA * 256;
  float* hs      = ws + o; o += NA * 256;
  float* tx1     = ws + o; o += NT * 256;   // later reused in-place as agg2
  float* a_s     = ws + o; o += NA * 4;
  float* a_d     = ws + o; o += NA * 4;
  unsigned* mEnc = (unsigned*)(ws + o); o += NA * 4;
  float* denom   = ws + o; o += NA * 4;
  float* elog    = ws + o; o += 550000LL * 4;
  float* wfs     = ws + o; o += 1024;
  float* wfd     = ws + o; o += 1024;
  // ---- workspace: f16 region ----
  _Float16* hb = (_Float16*)(ws + o);
  long long ho = 0;
  _Float16* x16   = hb + ho; ho += NT * 165;
  _Float16* xa16  = hb + ho; ho += NA * 64;
  _Float16* nt16  = hb + ho; ho += NT * 64;
  _Float16* na16  = hb + ho; ho += NA * 64;
  _Float16* h16t  = hb + ho; ho += NT * 64;
  _Float16* h16a  = hb + ho; ho += NA * 64;
  _Float16* nt1h  = hb + ho; ho += NT * 256;
  _Float16* na1h  = hb + ho; ho += NA * 256;
  _Float16* pTtx  = hb + ho; ho += 64LL * 165;
  _Float16* pTad  = hb + ho; ho += 64LL * 64;
  _Float16* rTtx  = hb + ho; ho += 256LL * 64;
  _Float16* rTad  = hb + ho; ho += 256LL * 64;
  _Float16* gT[6];
  for (int i = 0; i < 4; ++i) { gT[i] = hb + ho; ho += 256LL * 64; }   // layer-0 GATs
  for (int i = 4; i < 6; ++i) { gT[i] = hb + ho; ho += 256LL * 256; }  // layer-1 GATs

  float* logits = (float*)d_out;
  float* addr1  = (float*)d_out + NT * 2;
  float* agg2   = tx1;   // in-place reuse after layer-1 LN consumed tx1

  auto cdiv = [](long long a, long long b) { return (unsigned)((a + b - 1) / b); };

  auto gemm = [&](const _Float16* A, const _Float16* Bt, const float* bias, float* C,
                  long long M, int N, int K, bool accum, bool withBias) {
    dim3 g(cdiv(M, 32), (unsigned)(N / 64));
    if (accum)
      k_wmma_gemm<true, true><<<g, 256, 0, stream>>>(A, Bt, bias, C, (int)M, N, K);
    else if (withBias)
      k_wmma_gemm<false, true><<<g, 256, 0, stream>>>(A, Bt, bias, C, (int)M, N, K);
    else
      k_wmma_gemm<false, false><<<g, 256, 0, stream>>>(A, Bt, bias, C, (int)M, N, K);
  };

  auto run_gat = [&](const int* src, const int* dst, int E, int nloop,
                     const _Float16* xs, long long Ms, const _Float16* xd, long long Md,
                     int K, const Gat& g, const _Float16* WT, float* out_agg) {
    gemm(xs, WT, nullptr, hs, Ms, 256, K, false, false);        // hs = xs @ W
    k_fold_att<<<cdiv(K*4, 256), 256, 0, stream>>>(g.W, g.as, wfs, K);
    k_fold_att<<<cdiv(K*4, 256), 256, 0, stream>>>(g.W, g.ad, wfd, K);
    k_attn_gemv<<<cdiv(Ms, 8), 256, 0, stream>>>(xs, wfs, a_s, (int)Ms, K);
    k_attn_gemv<<<cdiv(Md, 8), 256, 0, stream>>>(xd, wfd, a_d, (int)Md, K);
    k_fill_u32<<<cdiv(Md*4, 256), 256, 0, stream>>>(mEnc, Md*4, 0u);
    k_fill_f32<<<cdiv(Md*4, 256), 256, 0, stream>>>(denom, Md*4, 0.f);
    int Et = E + nloop;
    k_edge_logit_max<<<cdiv(Et, 256), 256, 0, stream>>>(src, dst, E, nloop, a_s, a_d, elog, mEnc);
    k_edge_exp_sum<<<cdiv(Et, 256), 256, 0, stream>>>(dst, E, nloop, elog, mEnc, denom);
    k_edge_scatter<<<cdiv(Et, 8), 256, 0, stream>>>(src, dst, E, nloop, hs, elog, denom, out_agg);
  };

  // ---- one-time f16 conversions / weight transposes ----
  k_cvt_f16<<<cdiv(NT*165, 256), 256, 0, stream>>>(x_tx,   x16,  NT*165);
  k_cvt_f16<<<cdiv(NA*64, 256),  256, 0, stream>>>(x_addr, xa16, NA*64);
  k_transpose_cvt<<<cdiv(165LL*64, 256),  256, 0, stream>>>(proj_tx_W,   pTtx, 165, 64);
  k_transpose_cvt<<<cdiv(64LL*64, 256),   256, 0, stream>>>(proj_addr_W, pTad, 64,  64);
  k_transpose_cvt<<<cdiv(64LL*256, 256),  256, 0, stream>>>(res_tx_W,    rTtx, 64,  256);
  k_transpose_cvt<<<cdiv(64LL*256, 256),  256, 0, stream>>>(res_addr_W,  rTad, 64,  256);
  for (int i = 0; i < 4; ++i)
    k_transpose_cvt<<<cdiv(64LL*256, 256), 256, 0, stream>>>(gp[i].W, gT[i], 64, 256);
  for (int i = 4; i < 6; ++i)
    k_transpose_cvt<<<cdiv(256LL*256, 256), 256, 0, stream>>>(gp[i].W, gT[i], 256, 256);

  // ---- input projections ----
  gemm(x16,  pTtx, proj_tx_b,   h_tx,   NT, 64, 165, false, true);
  gemm(xa16, pTad, proj_addr_b, h_addr, NA, 64, 64,  false, true);

  // ---- layer 0 ----
  k_layernorm<64><<<cdiv(NT, 8), 256, 0, stream>>>(h_tx,   ln0_tx_g,   ln0_tx_b,   nt16, (int)NT);
  k_layernorm<64><<<cdiv(NA, 8), 256, 0, stream>>>(h_addr, ln0_addr_g, ln0_addr_b, na16, (int)NA);
  k_cvt_f16<<<cdiv(NT*64, 256), 256, 0, stream>>>(h_tx,   h16t, NT*64);
  k_cvt_f16<<<cdiv(NA*64, 256), 256, 0, stream>>>(h_addr, h16a, NA*64);
  k_rows_bias2<<<cdiv(NT*256, 256), 256, 0, stream>>>(agg_tx, tt0.bias, at0.bias, NT*256);
  k_rows_bias2<<<cdiv(NA*256, 256), 256, 0, stream>>>(agg_ad, aa0.bias, ta0.bias, NA*256);

  run_gat(edge_tt, edge_tt + E_TT, E_TT, (int)NT, nt16, NT, nt16, NT, 64, tt0, gT[0], agg_tx);
  run_gat(e_at_src, e_at_dst,      E_X,  0,       na16, NA, nt16, NT, 64, at0, gT[2], agg_tx);
  run_gat(edge_aa, edge_aa + E_AA, E_AA, (int)NA, na16, NA, na16, NA, 64, aa0, gT[1], agg_ad);
  run_gat(e_ta_src, e_ta_dst,      E_X,  0,       nt16, NT, na16, NA, 64, ta0, gT[3], agg_ad);

  gemm(h16t, rTtx, res_tx_b,   agg_tx, NT, 256, 64, true, true);   // += proj residual
  gemm(h16a, rTad, res_addr_b, agg_ad, NA, 256, 64, true, true);

  k_elu<<<cdiv(NT*256, 256), 256, 0, stream>>>(agg_tx, tx1,   NT*256);
  k_elu<<<cdiv(NA*256, 256), 256, 0, stream>>>(agg_ad, addr1, NA*256);   // output #2

  // ---- layer 1 (only tx updated) ----
  k_layernorm<256><<<cdiv(NT, 8), 256, 0, stream>>>(tx1,   ln1_tx_g,   ln1_tx_b,   nt1h, (int)NT);
  k_layernorm<256><<<cdiv(NA, 8), 256, 0, stream>>>(addr1, ln1_addr_g, ln1_addr_b, na1h, (int)NA);
  // identity residual + biases, in place: agg2(=tx1) += tt1.b + at1.b
  k_add_bias2<<<cdiv(NT*256, 256), 256, 0, stream>>>(agg2, tt1.bias, at1.bias, NT*256);

  run_gat(edge_tt, edge_tt + E_TT, E_TT, (int)NT, nt1h, NT, nt1h, NT, 256, tt1, gT[4], agg2);
  run_gat(e_at_src, e_at_dst,      E_X,  0,       na1h, NA, nt1h, NT, 256, at1, gT[5], agg2);

  // ---- fused ELU + classifier -> logits ----
  k_cls<<<cdiv(NT, 8), 256, 0, stream>>>(agg2, cls_W, cls_b, logits, (int)NT);
}
